// ClusterMemory_27814208209141
// MI455X (gfx1250) — matrix operand order repaired and checked by reading, not compiled, not run on hardware
//
#include <hip/hip_runtime.h>
#include <hip/hip_bf16.h>
#include <math.h>

// ---------------- problem constants (from reference) ----------------
#define DDIM   768        // feature dim (K)
#define BTOT   512        // 256 rgb rows + 256 ir rows (M)
#define NFEAT  65536      // memory bank rows (N)
#define SCALE  20.0f      // 1 / TEMP (TEMP = 0.05)

// ---------------- GEMM tiling ----------------
#define M_BLK  128
#define N_BLK  256
#define K_BLK  32
#define KTILES (DDIM / K_BLK)        // 24
#define NCHUNK (NFEAT / N_BLK)       // 256 column chunks
#define LDS_PAD 8
#define SROW   (K_BLK + LDS_PAD)     // 40 bf16 per LDS row (80B: dodges bank conflicts)

typedef __attribute__((ext_vector_type(16))) __bf16 v16bf;
typedef __attribute__((ext_vector_type(8)))  float  v8f;

union FragV { v16bf v; uint4 q[2]; };
union Pack32 { unsigned short u[32]; uint4 q[4]; };
union Pack8  { unsigned short u[8];  uint4 q;    };

// fp32 -> bf16 using the hardware converter (v_cvt path)
static __device__ __forceinline__ unsigned short f2bf(float f) {
    union { __bf16 b; unsigned short u; } c;
    c.b = (__bf16)f;
    return c.u;
}

static __device__ __forceinline__ unsigned lds_off(const void* p) {
    return (unsigned)(unsigned long long)p;   // LDS aperture: addr[31:0] == LDS offset
}

// =====================================================================
// Kernel 0: stream-convert features fp32 -> bf16 (read once from HBM;
// bf16 bank = 100 MB, fits the 192 MB L2 for the GEMM re-reads)
// =====================================================================
__global__ void k_convert_feat(const float* __restrict__ feat,
                               unsigned short* __restrict__ Fbf) {
    const size_t idx = ((size_t)blockIdx.x * 256 + threadIdx.x) * 8;
    const float4* s = (const float4*)(feat + idx);
    float4 a = s[0], b = s[1];
    Pack8 p;
    p.u[0] = f2bf(a.x); p.u[1] = f2bf(a.y); p.u[2] = f2bf(a.z); p.u[3] = f2bf(a.w);
    p.u[4] = f2bf(b.x); p.u[5] = f2bf(b.y); p.u[6] = f2bf(b.z); p.u[7] = f2bf(b.w);
    *(uint4*)(Fbf + idx) = p.q;
}

// =====================================================================
// Kernel 1: L2-normalize rows of [rgb; ir] and pack to bf16 A[512][768]
// =====================================================================
__global__ void k_normalize_pack(const float* __restrict__ rgb,
                                 const float* __restrict__ ir,
                                 unsigned short* __restrict__ Abf,
                                 float* __restrict__ inv_norm) {
    __shared__ float red[256];
    const int row = blockIdx.x;
    const int t   = threadIdx.x;
    const float* src = (row < 256) ? (rgb + (size_t)row * DDIM)
                                   : (ir  + (size_t)(row - 256) * DDIM);
    float x0 = src[t], x1 = src[t + 256], x2 = src[t + 512];
    red[t] = x0 * x0 + x1 * x1 + x2 * x2;
    __syncthreads();
    for (int s = 128; s > 0; s >>= 1) {
        if (t < s) red[t] += red[t + s];
        __syncthreads();
    }
    const float inv = 1.0f / fmaxf(sqrtf(red[0]), 1e-12f);
    if (t == 0) inv_norm[row] = inv;
    unsigned short* dst = Abf + (size_t)row * DDIM;
    dst[t]       = f2bf(x0 * inv);
    dst[t + 256] = f2bf(x1 * inv);
    dst[t + 512] = f2bf(x2 * inv);
}

// =====================================================================
// Shared epilogue for the GEMM kernels: fused per-row (max, sumexp)
// over this 256-column chunk, then per-chunk partial emit.
// C layout: VGPR r holds row = tile*16 + r + 8*(lane/16); col = lane%16.
// =====================================================================
#define GEMM_EPILOGUE()                                                        \
    _Pragma("unroll")                                                          \
    for (int i = 0; i < 4; ++i) {                                              \
        _Pragma("unroll")                                                      \
        for (int r = 0; r < 8; ++r) {                                          \
            float v0 = acc[i][0][r] * SCALE;                                   \
            float v1 = acc[i][1][r] * SCALE;                                   \
            float v2 = acc[i][2][r] * SCALE;                                   \
            float v3 = acc[i][3][r] * SCALE;                                   \
            float m = fmaxf(fmaxf(v0, v1), fmaxf(v2, v3));                     \
            float s = __expf(v0 - m) + __expf(v1 - m) +                        \
                      __expf(v2 - m) + __expf(v3 - m);                         \
            _Pragma("unroll")                                                  \
            for (int mask = 1; mask < 16; mask <<= 1) {                        \
                float om = __shfl_xor(m, mask, 16);                            \
                float os = __shfl_xor(s, mask, 16);                            \
                float nm = fmaxf(m, om);                                       \
                s = s * __expf(m - nm) + os * __expf(om - nm);                 \
                m = nm;                                                        \
            }                                                                  \
            if (l16 == 0) {                                                    \
                const int row_loc = wave_m * 64 + i * 16 + r + 8 * half;       \
                red_m[wave_n][row_loc] = m;                                    \
                red_s[wave_n][row_loc] = s;                                    \
            }                                                                  \
        }                                                                      \
    }                                                                          \
    __syncthreads();                                                           \
    if (t < M_BLK) {                                                           \
        float m = red_m[0][t], s = red_s[0][t];                                \
        _Pragma("unroll")                                                      \
        for (int w = 1; w < 4; ++w) {                                          \
            float om = red_m[w][t], os = red_s[w][t];                          \
            float nm = fmaxf(m, om);                                           \
            s = s * __expf(m - nm) + os * __expf(om - nm);                     \
            m = nm;                                                            \
        }                                                                      \
        const size_t row = (size_t)(mBase + t);                                \
        Pmax[row * NCHUNK + blockIdx.x] = m;                                   \
        Psum[row * NCHUNK + blockIdx.x] = s;                                   \
    }

// =====================================================================
// Kernel 2a (primary): BF16 WMMA GEMM, BOTH tiles staged with
// global_load_async_to_lds_b128 (ASYNCcnt), double-buffered LDS.
// Inner loop has zero conversion VALU.
// grid = (NCHUNK, BTOT/M_BLK), block = 256 threads (8 waves)
// =====================================================================
__global__ __launch_bounds__(256)
void k_gemm_bf(const unsigned short* __restrict__ Abf,
               const unsigned short* __restrict__ Fbf,
               float* __restrict__ Pmax,
               float* __restrict__ Psum) {
    __shared__ unsigned short sA[2][M_BLK][SROW];   // 2 x 10.0 KB
    __shared__ unsigned short sB[2][N_BLK][SROW];   // 2 x 20.0 KB
    __shared__ float red_m[4][M_BLK];
    __shared__ float red_s[4][M_BLK];

    const int t      = threadIdx.x;
    const int lane   = t & 31;
    const int wave   = t >> 5;
    const int wave_m = wave & 1;
    const int wave_n = wave >> 1;
    const int l16    = lane & 15;
    const int half   = lane >> 4;

    const int mBase = blockIdx.y * M_BLK;
    const int nBase = blockIdx.x * N_BLK;

    const int rA = t >> 1, hA = t & 1;              // A: 32B per thread
    const unsigned short* gA = Abf + (size_t)(mBase + rA) * DDIM + hA * 16;
    const unsigned short* gB = Fbf + (size_t)(nBase + t) * DDIM;   // B: 64B per thread

    v8f acc[4][4];
#pragma unroll
    for (int i = 0; i < 4; ++i)
#pragma unroll
        for (int j = 0; j < 4; ++j)
#pragma unroll
            for (int e = 0; e < 8; ++e) acc[i][j][e] = 0.0f;

    auto issueA = [&](int kb, int buf) {
        const unsigned lds = lds_off(&sA[buf][rA][hA * 16]);
        const unsigned long long ga = (unsigned long long)(gA + kb);
        asm volatile(
            "global_load_async_to_lds_b128 %0, %1, off\n\t"
            "global_load_async_to_lds_b128 %0, %1, off offset:16"
            :: "v"(lds), "v"(ga) : "memory");
    };
    auto issueB = [&](int kb, int buf) {
        const unsigned lds = lds_off(&sB[buf][t][0]);
        const unsigned long long gb = (unsigned long long)(gB + kb);
        asm volatile(
            "global_load_async_to_lds_b128 %0, %1, off\n\t"
            "global_load_async_to_lds_b128 %0, %1, off offset:16\n\t"
            "global_load_async_to_lds_b128 %0, %1, off offset:32\n\t"
            "global_load_async_to_lds_b128 %0, %1, off offset:48"
            :: "v"(lds), "v"(gb) : "memory");
    };

    // prologue: stage k-tile 0 into buffer 0
    issueA(0, 0);
    issueB(0, 0);
    asm volatile("s_wait_asynccnt 0x0" ::: "memory");
    __syncthreads();

    for (int kt = 0; kt < KTILES; ++kt) {
        const int  cur      = kt & 1;
        const int  kb_next  = (kt + 1) * K_BLK;
        const bool has_next = (kb_next < DDIM);

        if (has_next) {
            issueA(kb_next, cur ^ 1);
            issueB(kb_next, cur ^ 1);
            if (kb_next + K_BLK < DDIM)
                __builtin_prefetch(gB + kb_next + K_BLK, 0, 3);  // global_prefetch_b8
        }

        FragV bfr[4];
#pragma unroll
        for (int j = 0; j < 4; ++j) {
            const int n_loc = wave_n * 64 + j * 16 + l16;
            const uint4* p = (const uint4*)(&sB[cur][n_loc][half * 16]);
            bfr[j].q[0] = p[0];
            bfr[j].q[1] = p[1];
        }
#pragma unroll
        for (int i = 0; i < 4; ++i) {
            const int m_loc = wave_m * 64 + i * 16 + l16;
            FragV af;
            af.q[0] = *(const uint4*)(&sA[cur][m_loc][half * 16]);
            af.q[1] = *(const uint4*)(&sA[cur][m_loc][half * 16 + 8]);
#pragma unroll
            for (int j = 0; j < 4; ++j)
                acc[i][j] = __builtin_amdgcn_wmma_f32_16x16x32_bf16(
                    false, af.v, false, bfr[j].v, (short)0, acc[i][j], false, false);
        }

        asm volatile("s_wait_asynccnt 0x0" ::: "memory");
        __syncthreads();
    }

    GEMM_EPILOGUE()
}

// =====================================================================
// Kernel 2b (fallback, small workspace): B tile converted in-kernel.
// =====================================================================
__global__ __launch_bounds__(256)
void k_gemm_cvt(const unsigned short* __restrict__ Abf,
                const float* __restrict__ feat,
                float* __restrict__ Pmax,
                float* __restrict__ Psum) {
    __shared__ unsigned short sA[2][M_BLK][SROW];
    __shared__ unsigned short sB[2][N_BLK][SROW];
    __shared__ float red_m[4][M_BLK];
    __shared__ float red_s[4][M_BLK];

    const int t      = threadIdx.x;
    const int lane   = t & 31;
    const int wave   = t >> 5;
    const int wave_m = wave & 1;
    const int wave_n = wave >> 1;
    const int l16    = lane & 15;
    const int half   = lane >> 4;

    const int mBase = blockIdx.y * M_BLK;
    const int nBase = blockIdx.x * N_BLK;

    const int rA = t >> 1, hA = t & 1;
    const unsigned short* gA = Abf + (size_t)(mBase + rA) * DDIM + hA * 16;
    const float*          gB = feat + (size_t)(nBase + t) * DDIM;

    v8f acc[4][4];
#pragma unroll
    for (int i = 0; i < 4; ++i)
#pragma unroll
        for (int j = 0; j < 4; ++j)
#pragma unroll
            for (int e = 0; e < 8; ++e) acc[i][j][e] = 0.0f;

    auto issueA = [&](int kb, int buf) {
        const unsigned lds = lds_off(&sA[buf][rA][hA * 16]);
        const unsigned long long ga = (unsigned long long)(gA + kb);
        asm volatile(
            "global_load_async_to_lds_b128 %0, %1, off\n\t"
            "global_load_async_to_lds_b128 %0, %1, off offset:16"
            :: "v"(lds), "v"(ga) : "memory");
    };
    auto storeB = [&](const float4* breg, int buf) {
        Pack32 tmp;
#pragma unroll
        for (int q = 0; q < 8; ++q) {
            float4 v = breg[q];
            tmp.u[q * 4 + 0] = f2bf(v.x);
            tmp.u[q * 4 + 1] = f2bf(v.y);
            tmp.u[q * 4 + 2] = f2bf(v.z);
            tmp.u[q * 4 + 3] = f2bf(v.w);
        }
        uint4* dst = (uint4*)(&sB[buf][t][0]);
#pragma unroll
        for (int q = 0; q < 4; ++q) dst[q] = tmp.q[q];
    };

    issueA(0, 0);
    {
        const float4* s = (const float4*)gB;
        float4 breg[8];
#pragma unroll
        for (int q = 0; q < 8; ++q) breg[q] = s[q];
        storeB(breg, 0);
    }
    asm volatile("s_wait_asynccnt 0x0" ::: "memory");
    __syncthreads();

    for (int kt = 0; kt < KTILES; ++kt) {
        const int  cur      = kt & 1;
        const int  kb_next  = (kt + 1) * K_BLK;
        const bool has_next = (kb_next < DDIM);

        float4 breg[8];
        if (has_next) {
            issueA(kb_next, cur ^ 1);
            const float4* s = (const float4*)(gB + kb_next);
#pragma unroll
            for (int q = 0; q < 8; ++q) breg[q] = s[q];
            if (kb_next + K_BLK < DDIM)
                __builtin_prefetch(gB + kb_next + K_BLK, 0, 3);
        }

        FragV bfr[4];
#pragma unroll
        for (int j = 0; j < 4; ++j) {
            const int n_loc = wave_n * 64 + j * 16 + l16;
            const uint4* p = (const uint4*)(&sB[cur][n_loc][half * 16]);
            bfr[j].q[0] = p[0];
            bfr[j].q[1] = p[1];
        }
#pragma unroll
        for (int i = 0; i < 4; ++i) {
            const int m_loc = wave_m * 64 + i * 16 + l16;
            FragV af;
            af.q[0] = *(const uint4*)(&sA[cur][m_loc][half * 16]);
            af.q[1] = *(const uint4*)(&sA[cur][m_loc][half * 16 + 8]);
#pragma unroll
            for (int j = 0; j < 4; ++j)
                acc[i][j] = __builtin_amdgcn_wmma_f32_16x16x32_bf16(
                    false, af.v, false, bfr[j].v, (short)0, acc[i][j], false, false);
        }

        if (has_next) storeB(breg, cur ^ 1);

        asm volatile("s_wait_asynccnt 0x0" ::: "memory");
        __syncthreads();
    }

    GEMM_EPILOGUE()
}

// =====================================================================
// Kernel 3: per-row loss. Exact fp32 target logit (dot) + merge chunk
// partials -> loss_row = -(z_t - M - log S).  grid = 512, block = 256.
// =====================================================================
__global__ void k_row_loss(const float* __restrict__ rgb,
                           const float* __restrict__ ir,
                           const int* __restrict__ trgb,
                           const int* __restrict__ tir,
                           const float* __restrict__ feat,
                           const float* __restrict__ inv_norm,
                           const float* __restrict__ Pmax,
                           const float* __restrict__ Psum,
                           float* __restrict__ row_loss) {
    __shared__ float red[256];
    const int row = blockIdx.x;
    const int t   = threadIdx.x;
    const float* x = (row < 256) ? (rgb + (size_t)row * DDIM)
                                 : (ir  + (size_t)(row - 256) * DDIM);
    const int tgt = (row < 256) ? trgb[row] : tir[row - 256];
    const float* f = feat + (size_t)tgt * DDIM;

    red[t] = x[t] * f[t] + x[t + 256] * f[t + 256] + x[t + 512] * f[t + 512];
    __syncthreads();
    for (int s = 128; s > 0; s >>= 1) {
        if (t < s) red[t] += red[t + s];
        __syncthreads();
    }
    const float tgt_logit = red[0] * inv_norm[row] * SCALE;
    __syncthreads();

    const float m_c = Pmax[(size_t)row * NCHUNK + t];
    const float s_c = Psum[(size_t)row * NCHUNK + t];
    red[t] = m_c;
    __syncthreads();
    for (int s = 128; s > 0; s >>= 1) {
        if (t < s) red[t] = fmaxf(red[t], red[t + s]);
        __syncthreads();
    }
    const float M = red[0];
    __syncthreads();
    red[t] = s_c * __expf(m_c - M);
    __syncthreads();
    for (int s = 128; s > 0; s >>= 1) {
        if (t < s) red[t] += red[t + s];
        __syncthreads();
    }
    if (t == 0)
        row_loss[row] = -(tgt_logit - M - __logf(red[0]));
}

// =====================================================================
// Kernel 4: means over the two 256-row halves -> d_out[0..1]
// =====================================================================
__global__ void k_final(const float* __restrict__ row_loss, float* __restrict__ out) {
    __shared__ float r0[256];
    __shared__ float r1[256];
    const int t = threadIdx.x;
    r0[t] = row_loss[t];
    r1[t] = row_loss[256 + t];
    __syncthreads();
    for (int s = 128; s > 0; s >>= 1) {
        if (t < s) { r0[t] += r0[t + s]; r1[t] += r1[t + s]; }
        __syncthreads();
    }
    if (t == 0) {
        out[0] = r0[0] * (1.0f / 256.0f);
        out[1] = r1[0] * (1.0f / 256.0f);
    }
}

// =====================================================================
extern "C" void kernel_launch(void* const* d_in, const int* in_sizes, int n_in,
                              void* d_out, int out_size, void* d_ws, size_t ws_size,
                              hipStream_t stream) {
    const float* rgb  = (const float*)d_in[0];   // [256,768]
    const float* ir   = (const float*)d_in[1];   // [256,768]
    const int*   trgb = (const int*)d_in[2];     // [256]
    const int*   tir  = (const int*)d_in[3];     // [256]
    const float* feat = (const float*)d_in[4];   // [65536,768]
    float* out = (float*)d_out;                  // [2] fp32

    // workspace layout
    char* ws = (char*)d_ws;
    unsigned short* Abf   = (unsigned short*)(ws);                 // 512*768*2 = 786432
    float* inv_norm       = (float*)(ws + 786432);                 // 512*4     = 2048
    float* Pmax           = (float*)(ws + 788480);                 // 512*256*4 = 524288
    float* Psum           = (float*)(ws + 1312768);                // 512*256*4 = 524288
    float* row_loss       = (float*)(ws + 1837056);                // 512*4
    const size_t base_bytes = 1839104;
    const size_t fbf_bytes  = (size_t)NFEAT * DDIM * 2;            // 100663296
    unsigned short* Fbf   = (unsigned short*)(ws + base_bytes);

    k_normalize_pack<<<BTOT, 256, 0, stream>>>(rgb, ir, Abf, inv_norm);

    dim3 g2(NCHUNK, BTOT / M_BLK);   // 256 x 4
    if (ws_size >= base_bytes + fbf_bytes) {
        // primary path: bf16 feature bank (fits L2), fully async staging
        k_convert_feat<<<(NFEAT * (DDIM / 8)) / 256, 256, 0, stream>>>(feat, Fbf);
        k_gemm_bf<<<g2, 256, 0, stream>>>(Abf, Fbf, Pmax, Psum);
    } else {
        k_gemm_cvt<<<g2, 256, 0, stream>>>(Abf, feat, Pmax, Psum);
    }

    k_row_loss<<<BTOT, 256, 0, stream>>>(rgb, ir, trgb, tir, feat,
                                         inv_norm, Pmax, Psum, row_loss);

    k_final<<<1, 256, 0, stream>>>(row_loss, out);
}
